// EdgeConvFeatures_69784628625743
// MI455X (gfx1250) — matrix operand order repaired and checked by reading, not compiled, not run on hardware
//
#include <hip/hip_runtime.h>
#include <hip/hip_bf16.h>
#include <math.h>

typedef __attribute__((ext_vector_type(16))) _Float16 v16h;
typedef __attribute__((ext_vector_type(8)))  _Float16 v8h;
typedef __attribute__((ext_vector_type(8)))  float    v8f;

#define BATCH 16
#define KNN   10
#define HID   64

__device__ __forceinline__ v16h cat8(v8h lo, v8h hi) {
  return __builtin_shufflevector(lo, hi, 0, 1, 2, 3, 4, 5, 6, 7,
                                 8, 9, 10, 11, 12, 13, 14, 15);
}

// ---------------------------------------------------------------------------
// kNN: one wave per 16 query rows of one batch. Input is f16, zero-padded to
// LD columns, so all fragment loads are unconditional 16B vector loads.
// Lanes 0-15 keep a sorted top-10 (ascending distance, index tie-break).
// ---------------------------------------------------------------------------
template <int LD>
__global__ void knn_kernel(const _Float16* __restrict__ x16, int N,
                           int* __restrict__ knn_idx) {
  __shared__ float rowsq_s[16];
  __shared__ float colsq_s[16];
  __shared__ float tile_s[16 * 16];

  const int b    = blockIdx.y;
  const int q0   = blockIdx.x * 16;
  const int lane = threadIdx.x;
  const int mn   = lane & 15;
  const int koff = (lane < 16) ? 0 : 8;   // A-frag K base (halves)
  const int kb   = (lane < 16) ? 0 : 8;   // B-frag K base (halves)
  const _Float16* xb = x16 + (size_t)b * N * LD;
  const _Float16* ar = xb + (size_t)(q0 + mn) * LD;

  if (lane < 16) {
    float s = 0.f;
#pragma unroll
    for (int c = 0; c < LD; ++c) { float v = (float)ar[c]; s += v * v; }
    rowsq_s[mn] = s;
  }

  float bd[KNN]; int bi[KNN];
#pragma unroll
  for (int s = 0; s < KNN; ++s) { bd[s] = 3.0e38f; bi[s] = 0x7fffffff; }

  for (int j0 = 0; j0 < N; j0 += 16) {
    __syncthreads();                       // protect LDS reuse across iterations
    const _Float16* bc = xb + (size_t)(j0 + mn) * LD;
    if (lane < 16) {
      float s = 0.f;
#pragma unroll
      for (int c = 0; c < LD; ++c) { float v = (float)bc[c]; s += v * v; }
      colsq_s[mn] = s;
    }

    v8f acc = {};
#pragma unroll
    for (int kc = 0; kc < LD; kc += 32) {
      v16h af = cat8(*(const v8h*)(ar + kc + koff),
                     *(const v8h*)(ar + kc + 16 + koff));
      v16h bf = cat8(*(const v8h*)(bc + kc + kb),
                     *(const v8h*)(bc + kc + kb + 16));
      acc = __builtin_amdgcn_wmma_f32_16x16x32_f16(
          false, af, false, bf, (short)0, acc, false, false);
    }
    __syncthreads();                       // colsq/rowsq visible

    const int rb  = (lane < 16) ? 0 : 8;
    float     csq = colsq_s[mn];
#pragma unroll
    for (int r = 0; r < 8; ++r) {
      tile_s[(rb + r) * 16 + mn] = rowsq_s[rb + r] + csq - 2.0f * acc[r];
    }
    __syncthreads();                       // tile visible for selection

    if (lane < 16) {
      for (int t = 0; t < 16; ++t) {
        float cd = tile_s[lane * 16 + t];
        int   ci = j0 + t;
#pragma unroll
        for (int s = 0; s < KNN; ++s) {
          bool ins = (cd < bd[s]) || (cd == bd[s] && ci < bi[s]);
          float td = bd[s]; int ti = bi[s];
          if (ins) { bd[s] = cd; bi[s] = ci; cd = td; ci = ti; }
        }
      }
    }
  }

  if (lane < 16) {
    int* op = knn_idx + ((size_t)b * N + q0 + lane) * KNN;
#pragma unroll
    for (int s = 0; s < KNN; ++s) op[s] = bi[s];
  }
}

// positions (f32, C cols) -> padded f16 [total, LD]
__global__ void pos_to_f16(const float* __restrict__ x, _Float16* __restrict__ x16,
                           int total, int C, int LD) {
  size_t tot = (size_t)total * LD;
  for (size_t t = (size_t)blockIdx.x * blockDim.x + threadIdx.x; t < tot;
       t += (size_t)gridDim.x * blockDim.x) {
    int c = (int)(t % LD);
    x16[t] = (c < C) ? (_Float16)x[(t / LD) * C + c] : (_Float16)0.f;
  }
}

// weights W[Kc,64] -> transposed, zero-padded f16 Wt[64,LDA]
__global__ void prep_w(const float* __restrict__ W, _Float16* __restrict__ Wt,
                       int Kc, int LDA) {
  int tot = HID * LDA;
  for (int t = blockIdx.x * blockDim.x + threadIdx.x; t < tot;
       t += gridDim.x * blockDim.x) {
    int n = t / LDA, k = t % LDA;
    Wt[t] = (k < Kc) ? (_Float16)W[k * HID + n] : (_Float16)0.f;
  }
}

// Wo[192,512] -> transposed f16 Wot[512,192]
__global__ void prep_wo(const float* __restrict__ Wo, _Float16* __restrict__ Wot) {
  int tot = 512 * 192;
  for (int t = blockIdx.x * blockDim.x + threadIdx.x; t < tot;
       t += gridDim.x * blockDim.x) {
    int n = t / 192, k = t % 192;
    Wot[t] = (_Float16)Wo[k * 512 + n];
  }
}

__global__ void feats_to_f16(const float* __restrict__ F, _Float16* __restrict__ F16) {
  int t = blockIdx.x * blockDim.x + threadIdx.x;
  if (t < BATCH * 192) F16[t] = (_Float16)F[t];
}

// ---------------------------------------------------------------------------
// Build edge features [xi, xj - xi] as f16, zero-padded to ldA columns.
// ---------------------------------------------------------------------------
__global__ void gather_edges(const float* __restrict__ x, const int* __restrict__ idx,
                             _Float16* __restrict__ eA, int N, int C, int ldA) {
  size_t total = (size_t)BATCH * N * KNN * ldA;
  for (size_t t = (size_t)blockIdx.x * blockDim.x + threadIdx.x; t < total;
       t += (size_t)gridDim.x * blockDim.x) {
    int    c  = (int)(t % ldA);
    size_t e  = t / ldA;          // edge id = (b*N+n)*KNN + k
    size_t bn = e / KNN;          // b*N + n
    const float* xi = x + bn * C;
    float v = 0.f;
    if (c < C) {
      v = xi[c];
    } else if (c < 2 * C) {
      int    j = idx[e];
      size_t b = bn / N;
      v = x[((size_t)b * N + j) * C + (c - C)] - xi[c - C];
    }
    eA[t] = (_Float16)v;
  }
}

// ---------------------------------------------------------------------------
// Edge MLP layer: H[E,64] = ReLU(A[E,LDA] @ Wt^T + bias). One wave per
// 16-row tile, 4 WMMA N-tiles, LDA/32 K-chunks (all compile-time).
// Accumulates per-channel sum/sumsq for BatchNorm.
// ---------------------------------------------------------------------------
template <int LDA>
__global__ void edge_gemm(const _Float16* __restrict__ A,
                          const _Float16* __restrict__ Wt,  // [64][LDA] padded
                          const float* __restrict__ bias,
                          float* __restrict__ H, float* __restrict__ stats, int E) {
  const int lane = threadIdx.x;
  const int mn   = lane & 15;
  const int koff = (lane < 16) ? 0 : 8;
  const int kb   = (lane < 16) ? 0 : 8;
  const int m0   = blockIdx.x * 16;
  const _Float16* ar = A + (size_t)(m0 + mn) * LDA;

  v8f acc[4] = {};
#pragma unroll
  for (int k0 = 0; k0 < LDA; k0 += 32) {
    v16h af = cat8(*(const v8h*)(ar + k0 + koff),
                   *(const v8h*)(ar + k0 + 16 + koff));
#pragma unroll
    for (int nt = 0; nt < 4; ++nt) {
      const _Float16* wr = Wt + (size_t)(nt * 16 + mn) * LDA;
      v16h bf = cat8(*(const v8h*)(wr + k0 + kb),
                     *(const v8h*)(wr + k0 + kb + 16));
      acc[nt] = __builtin_amdgcn_wmma_f32_16x16x32_f16(
          false, af, false, bf, (short)0, acc[nt], false, false);
    }
  }

  const int rb = (lane < 16) ? 0 : 8;
#pragma unroll
  for (int nt = 0; nt < 4; ++nt) {
    int   n  = nt * 16 + mn;
    float bv = bias[n];
    float s1 = 0.f, s2 = 0.f;
#pragma unroll
    for (int r = 0; r < 8; ++r) {
      int   row = m0 + rb + r;
      float v   = fmaxf(acc[nt][r] + bv, 0.f);
      H[(size_t)row * HID + n] = v;
      s1 += v; s2 += v * v;
    }
    atomicAdd(&stats[n], s1);
    atomicAdd(&stats[HID + n], s2);
  }
}

// ---------------------------------------------------------------------------
// BatchNorm (training batch stats) applied to H, emitted as f16 for next GEMM.
// ---------------------------------------------------------------------------
__global__ void bn_apply(const float* __restrict__ H, const float* __restrict__ stats,
                         float cntInv, const float* __restrict__ g,
                         const float* __restrict__ be, _Float16* __restrict__ out,
                         int E) {
  size_t total = (size_t)E * HID;
  for (size_t t = (size_t)blockIdx.x * blockDim.x + threadIdx.x; t < total;
       t += (size_t)gridDim.x * blockDim.x) {
    int   c    = (int)(t & (HID - 1));
    float mean = stats[c] * cntInv;
    float var  = stats[HID + c] * cntInv - mean * mean;
    float v    = (H[t] - mean) * rsqrtf(var + 1e-5f) * g[c] + be[c];
    out[t] = (_Float16)v;
  }
}

// max over k neighbors -> node features (f32)
__global__ void max_over_k(const _Float16* __restrict__ h, float* __restrict__ xo,
                           int N) {
  size_t total = (size_t)BATCH * N * HID;
  for (size_t t = (size_t)blockIdx.x * blockDim.x + threadIdx.x; t < total;
       t += (size_t)gridDim.x * blockDim.x) {
    int    c  = (int)(t & (HID - 1));
    size_t bn = t >> 6;
    const _Float16* hp = h + (bn * KNN) * HID + c;
    float m = -3.0e38f;
#pragma unroll
    for (int k = 0; k < KNN; ++k) m = fmaxf(m, (float)hp[k * HID]);
    xo[t] = m;
  }
}

// ---------------------------------------------------------------------------
// TopKPooling: score = tanh(x.w/||w||); LDS bitonic sort (desc, idx tie asc);
// keep top N/2 in rank order, scale by score. Writes f32 and padded f16 copy
// (consumed by next stage's kNN). One block per batch.
// ---------------------------------------------------------------------------
__global__ void topk_pool(const float* __restrict__ x, const float* __restrict__ w,
                          float* __restrict__ xo, _Float16* __restrict__ xo16,
                          int N) {
  __shared__ float sc[2048];
  __shared__ int   si[2048];
  const int b   = blockIdx.x;
  const int tid = threadIdx.x;

  float wn = 0.f;
#pragma unroll
  for (int c = 0; c < HID; ++c) { float v = w[c]; wn += v * v; }
  wn = sqrtf(wn);

  for (int n = tid; n < N; n += blockDim.x) {
    const float* xr = x + ((size_t)b * N + n) * HID;
    float dot = 0.f;
#pragma unroll
    for (int c = 0; c < HID; ++c) dot += xr[c] * w[c];
    sc[n] = tanhf(dot / wn);
    si[n] = n;
  }
  __syncthreads();

  for (int ksz = 2; ksz <= N; ksz <<= 1) {
    for (int j = ksz >> 1; j > 0; j >>= 1) {
      for (int i = tid; i < N; i += blockDim.x) {
        int ixj = i ^ j;
        if (ixj > i) {
          float a = sc[i], c2 = sc[ixj];
          int  ia = si[i], ib = si[ixj];
          bool afirst = (a > c2) || (a == c2 && ia < ib);
          bool desc   = ((i & ksz) == 0);
          if (desc ? !afirst : afirst) {
            sc[i] = c2; sc[ixj] = a; si[i] = ib; si[ixj] = ia;
          }
        }
      }
      __syncthreads();
    }
  }

  int kp = N >> 1;
  for (int t = tid; t < kp * HID; t += blockDim.x) {
    int r = t >> 6, c = t & (HID - 1);
    float v = x[((size_t)b * N + si[r]) * HID + c] * sc[r];
    size_t o = ((size_t)b * kp + r) * HID + c;
    xo[o]   = v;
    xo16[o] = (_Float16)v;
  }
}

// global max pool over nodes -> feats[b, foff + c]
__global__ void gmax(const float* __restrict__ xp, int Np,
                     float* __restrict__ feats, int foff) {
  int b = blockIdx.x, c = threadIdx.x;
  float m = -3.0e38f;
  for (int r = 0; r < Np; ++r)
    m = fmaxf(m, xp[((size_t)b * Np + r) * HID + c]);
  feats[b * (3 * HID) + foff + c] = m;
}

// out[16,512] = F16[16,192] @ Wot^T + bo. One WMMA M-tile (M=16), 32 N-blocks.
__global__ void out_gemm(const _Float16* __restrict__ F16,
                         const _Float16* __restrict__ Wot,  // [512][192]
                         const float* __restrict__ bo, float* __restrict__ out) {
  const int lane = threadIdx.x;
  const int mn   = lane & 15;
  const int koff = (lane < 16) ? 0 : 8;
  const int kb   = (lane < 16) ? 0 : 8;
  const int n    = blockIdx.x * 16 + mn;
  const _Float16* fr = F16 + mn * 192;
  const _Float16* wr = Wot + (size_t)n * 192;

  v8f acc = {};
#pragma unroll
  for (int k0 = 0; k0 < 192; k0 += 32) {
    v16h af = cat8(*(const v8h*)(fr + k0 + koff),
                   *(const v8h*)(fr + k0 + 16 + koff));
    v16h bf = cat8(*(const v8h*)(wr + k0 + kb),
                   *(const v8h*)(wr + k0 + kb + 16));
    acc = __builtin_amdgcn_wmma_f32_16x16x32_f16(
        false, af, false, bf, (short)0, acc, false, false);
  }
  const int rb = (lane < 16) ? 0 : 8;
  float bv = bo[n];
#pragma unroll
  for (int r = 0; r < 8; ++r) out[(rb + r) * 512 + n] = acc[r] + bv;
}

// ---------------------------------------------------------------------------
extern "C" void kernel_launch(void* const* d_in, const int* in_sizes, int n_in,
                              void* d_out, int out_size, void* d_ws, size_t ws_size,
                              hipStream_t stream) {
  (void)in_sizes; (void)n_in; (void)out_size; (void)ws_size;
  const float* positions = (const float*)d_in[0];
  auto P = [&](int i) { return (const float*)d_in[i]; };
  // d_in layout: positions, then 3 convs x 3 layers x (W,b,g,be), 3 pools, Wo, bo
  const float* poolw[3] = { P(37), P(38), P(39) };
  const float* Wo = P(40);
  const float* bo = P(41);

  char* ws = (char*)d_ws;
  size_t off = 0;
  auto take = [&](size_t bytes) -> char* {
    char* p = ws + off; off += (bytes + 255) & ~(size_t)255; return p;
  };
  const size_t Emax = (size_t)BATCH * 2048 * KNN;           // 327,680 edges
  _Float16* eA0   = (_Float16*)take(Emax * 128 * sizeof(_Float16));
  _Float16* eA1   = (_Float16*)take(Emax * HID * sizeof(_Float16));
  float*    Hbuf  = (float*)   take(Emax * HID * sizeof(float));
  float*    stats = (float*)   take(2 * HID * sizeof(float));
  float*    xa    = (float*)   take((size_t)BATCH * 2048 * HID * sizeof(float));
  float*    xb    = (float*)   take((size_t)BATCH * 2048 * HID * sizeof(float));
  _Float16* x16   = (_Float16*)take((size_t)BATCH * 2048 * HID * sizeof(_Float16));
  int*      knnI  = (int*)     take(Emax * sizeof(int));
  float*    feats = (float*)   take((size_t)BATCH * 3 * HID * sizeof(float));
  _Float16* F16   = (_Float16*)take((size_t)BATCH * 3 * HID * sizeof(_Float16));
  _Float16* Wt    = (_Float16*)take((size_t)HID * 128 * sizeof(_Float16));
  _Float16* Wot   = (_Float16*)take((size_t)512 * 192 * sizeof(_Float16));

  // f16 copy of positions, padded to 32 cols
  pos_to_f16<<<512, 256, 0, stream>>>(positions, x16, BATCH * 2048, 3, 32);

  const float* xin = positions;
  int N = 2048;
  for (int c = 0; c < 3; ++c) {
    const int C    = (c == 0) ? 3 : HID;
    const int twoC = 2 * C;
    const int ldA0 = (c == 0) ? 32 : 128;
    const int E    = BATCH * N * KNN;

    if (c == 0)
      knn_kernel<32><<<dim3(N / 16, BATCH), 32, 0, stream>>>(x16, N, knnI);
    else
      knn_kernel<64><<<dim3(N / 16, BATCH), 32, 0, stream>>>(x16, N, knnI);
    gather_edges<<<1024, 256, 0, stream>>>(xin, knnI, eA0, N, C, ldA0);

    for (int l = 0; l < 3; ++l) {
      const float* Wl  = P(1 + (c * 3 + l) * 4 + 0);
      const float* bl  = P(1 + (c * 3 + l) * 4 + 1);
      const float* gl  = P(1 + (c * 3 + l) * 4 + 2);
      const float* bel = P(1 + (c * 3 + l) * 4 + 3);
      const _Float16* Ain  = (l == 1) ? eA1 : eA0;
      _Float16*       Aout = (l == 1) ? eA0 : eA1;
      const int ldl = (l == 0) ? ldA0 : HID;
      const int Kcl = (l == 0) ? twoC : HID;

      prep_w<<<32, 256, 0, stream>>>(Wl, Wt, Kcl, ldl);
      (void)hipMemsetAsync(stats, 0, 2 * HID * sizeof(float), stream);
      if (ldl == 32)
        edge_gemm<32><<<E / 16, 32, 0, stream>>>(Ain, Wt, bl, Hbuf, stats, E);
      else if (ldl == 64)
        edge_gemm<64><<<E / 16, 32, 0, stream>>>(Ain, Wt, bl, Hbuf, stats, E);
      else
        edge_gemm<128><<<E / 16, 32, 0, stream>>>(Ain, Wt, bl, Hbuf, stats, E);
      bn_apply<<<1024, 256, 0, stream>>>(Hbuf, stats, 1.0f / (float)E, gl, bel,
                                         Aout, E);
    }
    max_over_k<<<1024, 256, 0, stream>>>(eA1, xa, N);
    topk_pool<<<BATCH, 256, 0, stream>>>(xa, poolw[c], xb, x16, N);
    N >>= 1;
    gmax<<<BATCH, HID, 0, stream>>>(xb, N, feats, c * HID);
    xin = xb;
  }

  feats_to_f16<<<12, 256, 0, stream>>>(feats, F16);
  prep_wo<<<384, 256, 0, stream>>>(Wo, Wot);
  out_gemm<<<32, 32, 0, stream>>>(F16, Wot, bo, (float*)d_out);
}
